// InversionTemporalMHA_60112362275006
// MI455X (gfx1250) — matrix operand order repaired and checked by reading, not compiled, not run on hardware
//
#include <hip/hip_runtime.h>
#include <hip/hip_bf16.h>
#include <stdint.h>

typedef __bf16 bf16;
typedef __attribute__((ext_vector_type(16))) __bf16 v16bf;
typedef __attribute__((ext_vector_type(8)))  float  v8f;

#define T_    128
#define BJ_   768
#define D_    512
#define NH_   8
#define HD_   64
#define RANK_ 16
#define NJ_   24
#define SCALING_ 2.0f
#define M_ROWS (T_*BJ_)   // 98304

union FragU { uint4 u[2]; v16bf v; };
union H8   { uint4 u; bf16 e[8]; };
union H4   { uint2 u; bf16 e[4]; };

// ---- CDNA5 async global->LDS copy (ASYNCcnt path), GVS addressing --------
__device__ __forceinline__ void async_ld128(unsigned lds_off, const void* base,
                                            unsigned byte_off) {
    // global_load_async_to_lds_b128  vdst(LDS addr), vaddr(32b offset), saddr(base)
    asm volatile("global_load_async_to_lds_b128 %0, %1, %2"
                 :: "v"(lds_off), "v"(byte_off), "s"(base)
                 : "memory");
}
__device__ __forceinline__ void wait_async0() {
    asm volatile("s_wait_asynccnt 0x0" ::: "memory");
}
__device__ __forceinline__ unsigned lds_off_of(const void* p) {
    // low 32 bits of a generic pointer into LDS == DS-space byte offset
    return (unsigned)(uintptr_t)p;
}

// ---------------------------------------------------------------------------
// Kernel 1: effective weights  W_eff = W + 2.0 * (B @ A), cast to bf16.
// ---------------------------------------------------------------------------
__global__ void __launch_bounds__(256)
prep_weights_kernel(const float* __restrict__ ipw, const float* __restrict__ out_w,
                    const float* __restrict__ Aq, const float* __restrict__ Bq,
                    const float* __restrict__ Ak, const float* __restrict__ Bk,
                    const float* __restrict__ Av, const float* __restrict__ Bv,
                    const float* __restrict__ Ao, const float* __restrict__ Bo,
                    bf16* __restrict__ Wout)
{
    const int which = blockIdx.y;
    const int g = blockIdx.x * 256 + threadIdx.x;   // 0..262143
    const int o = g >> 9, d = g & 511;
    const float* Wb = (which == 3) ? out_w : (ipw + (size_t)which * D_ * D_);
    const float* Am = (which == 0) ? Aq : (which == 1) ? Ak : (which == 2) ? Av : Ao;
    const float* Bm = (which == 0) ? Bq : (which == 1) ? Bk : (which == 2) ? Bv : Bo;
    float acc = 0.f;
#pragma unroll
    for (int r = 0; r < RANK_; ++r)
        acc += Bm[o * RANK_ + r] * Am[r * D_ + d];
    float val = Wb[(size_t)o * D_ + d] + SCALING_ * acc;
    Wout[(size_t)which * D_ * D_ + g] = (bf16)val;
}

// ---------------------------------------------------------------------------
// Kernel 2: Xq = bf16(query + m_exp), Xk = bf16(key + m_exp), Xv = bf16(value)
// ---------------------------------------------------------------------------
__global__ void __launch_bounds__(256)
prep_inputs_kernel(const float4* __restrict__ q4, const float4* __restrict__ k4,
                   const float4* __restrict__ v4, const float4* __restrict__ m4,
                   bf16* __restrict__ Xq, bf16* __restrict__ Xk, bf16* __restrict__ Xv)
{
    const size_t i4 = (size_t)blockIdx.x * 256 + threadIdx.x;  // < 12582912
    const int r  = (int)(i4 >> 7);
    const int c  = (int)(i4 & 127);
    const int t  = r / BJ_;
    const int bj = r - t * BJ_;
    const int b  = bj / NJ_;
    const float4 mv = m4[((size_t)b * T_ + t) * 128 + c];
    const float4 qv = q4[i4];
    const float4 kv = k4[i4];
    const float4 vv = v4[i4];
    H4 hq, hk, hv;
    hq.e[0] = (bf16)(qv.x + mv.x); hq.e[1] = (bf16)(qv.y + mv.y);
    hq.e[2] = (bf16)(qv.z + mv.z); hq.e[3] = (bf16)(qv.w + mv.w);
    hk.e[0] = (bf16)(kv.x + mv.x); hk.e[1] = (bf16)(kv.y + mv.y);
    hk.e[2] = (bf16)(kv.z + mv.z); hk.e[3] = (bf16)(kv.w + mv.w);
    hv.e[0] = (bf16)vv.x; hv.e[1] = (bf16)vv.y;
    hv.e[2] = (bf16)vv.z; hv.e[3] = (bf16)vv.w;
    *(uint2*)(Xq + i4 * 4) = hq.u;
    *(uint2*)(Xk + i4 * 4) = hk.u;
    *(uint2*)(Xv + i4 * 4) = hv.u;
}

// ---------------------------------------------------------------------------
// Kernel 3: Y[m,n] = sum_k A[m,k]*W[n,k] + bias[n]  (bf16 in, f32 accum)
// Tiles copied global->LDS with async b128 (no VGPR staging -> no spills),
// double-buffered, one barrier per K-step.
// ---------------------------------------------------------------------------
#define GBM 128
#define GBN 128
#define GBK 64
#define LROW 144     // 64 bf16 = 128B data + 16B pad
#define TILEB (GBM * LROW)          // 18432
#define BUFB  (2 * TILEB)           // 36864

__global__ void __launch_bounds__(256)
gemm_bf16_kernel(const bf16* __restrict__ A, const bf16* __restrict__ W,
                 const float* __restrict__ bias,
                 bf16* __restrict__ outb, float* __restrict__ outf)
{
    __shared__ __align__(16) char smem[2 * BUFB];   // 73728 B

    const int tid  = threadIdx.x;
    const int lane = tid & 31;
    const int w    = tid >> 5;
    const int wm   = w >> 2, wn = w & 3;
    const int h    = lane >> 4, lm = lane & 15;
    const int m0   = blockIdx.y * GBM;
    const int n0   = blockIdx.x * GBN;
    const int lrow = tid >> 3;   // 0..31
    const int lcol = tid & 7;    // 16B chunk within 128B row

    const unsigned sbase = lds_off_of(smem);

    v8f acc[4][2];
#pragma unroll
    for (int i = 0; i < 4; ++i)
#pragma unroll
        for (int j = 0; j < 2; ++j) acc[i][j] = {};

    auto issue = [&](int buf, int kt) {
        const unsigned kb = (unsigned)kt * (GBK * 2);   // k byte offset
#pragma unroll
        for (int i = 0; i < 4; ++i) {
            const int r = lrow + i * 32;
            const unsigned lo = sbase + buf * BUFB + r * LROW + lcol * 16;
            async_ld128(lo,         A, (unsigned)(m0 + r) * (D_ * 2) + kb + lcol * 16);
            async_ld128(lo + TILEB, W, (unsigned)(n0 + r) * (D_ * 2) + kb + lcol * 16);
        }
    };
    auto compute = [&](int buf) {
        const char* a = smem + buf * BUFB;
        const char* b = a + TILEB;
#pragma unroll
        for (int kc = 0; kc < 2; ++kc) {
            FragU af[4], bw[2];
#pragma unroll
            for (int mt = 0; mt < 4; ++mt) {
                const char* p = a + (wm * 64 + mt * 16 + lm) * LROW + kc * 64 + h * 16;
                af[mt].u[0] = *(const uint4*)p;
                af[mt].u[1] = *(const uint4*)(p + 32);
            }
#pragma unroll
            for (int nt = 0; nt < 2; ++nt) {
                const char* p = b + (wn * 32 + nt * 16 + lm) * LROW + kc * 64 + h * 16;
                bw[nt].u[0] = *(const uint4*)p;
                bw[nt].u[1] = *(const uint4*)(p + 32);
            }
#pragma unroll
            for (int mt = 0; mt < 4; ++mt)
#pragma unroll
                for (int nt = 0; nt < 2; ++nt)
                    acc[mt][nt] = __builtin_amdgcn_wmma_f32_16x16x32_bf16(
                        false, af[mt].v, false, bw[nt].v, (short)0, acc[mt][nt],
                        false, false);
        }
    };

    issue(0, 0);
    wait_async0();
    __syncthreads();
    const int NK = D_ / GBK;  // 8
    for (int kt = 0; kt < NK; ++kt) {
        if (kt + 1 < NK) issue((kt + 1) & 1, kt + 1);
        compute(kt & 1);
        if (kt + 1 < NK) wait_async0();
        __syncthreads();
    }

#pragma unroll
    for (int nt = 0; nt < 2; ++nt) {
        const int col = n0 + wn * 32 + nt * 16 + lm;
        const float bv = bias[col];
#pragma unroll
        for (int mt = 0; mt < 4; ++mt) {
#pragma unroll
            for (int j = 0; j < 8; ++j) {
                const int row = m0 + wm * 64 + mt * 16 + h * 8 + j;
                const float v = acc[mt][nt][j] + bv;
                if (outb) outb[(size_t)row * D_ + col] = (bf16)v;
                else      outf[(size_t)row * D_ + col] = v;
            }
        }
    }
}

// ---------------------------------------------------------------------------
// Kernel 4: per-(bj,head) attention, 128x128x64.
// K staged via async global->LDS; V staged manually (fused shift-diff +
// transpose); softmax unnormalized with 1/rowsum folded into PV epilogue.
// ---------------------------------------------------------------------------
__global__ void __launch_bounds__(256)
attn_kernel(const bf16* __restrict__ Q, const bf16* __restrict__ K,
            const bf16* __restrict__ V, bf16* __restrict__ O)
{
    __shared__ __align__(16) char smem[18432 + 17408 + 34816];
    char* sK = smem;              // 128 rows (t) x 144 B (64 bf16 + pad)
    char* sV = smem + 18432;      // transposed: 64 rows (d) x 272 B
    char* sP = sV + 17408;        // 8 waves x 16 rows x 272 B

    const int tid  = threadIdx.x;
    const int lane = tid & 31;
    const int w    = tid >> 5;
    const int h    = lane >> 4, lm = lane & 15;
    const int bj   = blockIdx.x >> 3;
    const int hd   = blockIdx.x & 7;
    const size_t base    = (size_t)bj * D_ + (size_t)hd * HD_;
    const size_t tstride = (size_t)BJ_ * D_;   // 393216 elems

    const bf16* Kp = K + base;
    const bf16* Vp = V + base;
    const unsigned sKoff = lds_off_of(sK);

    // --- stage K via async copies; V' manually (shift-diff + transpose) ---
#pragma unroll
    for (int i = 0; i < 4; ++i) {
        const int cidx = i * 256 + tid;
        const int r = cidx >> 3, cc = cidx & 7;
        async_ld128(sKoff + r * 144 + cc * 16, Kp,
                    (unsigned)(r * (BJ_ * D_ * 2)) + cc * 16);
        const bf16* vp = Vp + (size_t)r * tstride + cc * 8;
        H8 v0, v1;
        v0.u = *(const uint4*)vp;
        v1.u = (r < T_ - 1) ? *(const uint4*)(vp + tstride) : v0.u;
#pragma unroll
        for (int e = 0; e < 8; ++e) {
            const float dv = (float)v0.e[e] - (float)v1.e[e];
            const int dloc = cc * 8 + e;
            *(bf16*)(sV + dloc * 272 + r * 2) = (bf16)dv;
        }
    }

    // --- Q fragments (A layout) straight from global ---
    FragU aq[2];
#pragma unroll
    for (int kq = 0; kq < 2; ++kq) {
        const bf16* qp = Q + base + (size_t)(w * 16 + lm) * tstride + kq * 32 + h * 8;
        aq[kq].u[0] = *(const uint4*)qp;
        aq[kq].u[1] = *(const uint4*)(qp + 16);
    }
    wait_async0();
    __syncthreads();

    // --- S = Q K^T (16x128 per wave) ---
    v8f s[8];
#pragma unroll
    for (int nt = 0; nt < 8; ++nt) s[nt] = {};
#pragma unroll
    for (int kq = 0; kq < 2; ++kq) {
#pragma unroll
        for (int nt = 0; nt < 8; ++nt) {
            FragU bk;
            const char* p = sK + (nt * 16 + lm) * 144 + kq * 64 + h * 16;
            bk.u[0] = *(const uint4*)p;
            bk.u[1] = *(const uint4*)(p + 32);
            s[nt] = __builtin_amdgcn_wmma_f32_16x16x32_bf16(
                false, aq[kq].v, false, bk.v, (short)0, s[nt], false, false);
        }
    }

    // --- row softmax (unnormalized); wave halves hold distinct rows ---
    const float c_scale = 0.125f * 1.44269504088896340736f;  // scale*log2(e)
    float rsum[8];
#pragma unroll
    for (int j = 0; j < 8; ++j) {
        float mx = -3.0e38f;
#pragma unroll
        for (int nt = 0; nt < 8; ++nt) mx = fmaxf(mx, s[nt][j]);
#pragma unroll
        for (int o = 1; o < 16; o <<= 1) mx = fmaxf(mx, __shfl_xor(mx, o));
        float sm = 0.f;
#pragma unroll
        for (int nt = 0; nt < 8; ++nt) {
            const float p = exp2f((s[nt][j] - mx) * c_scale);
            s[nt][j] = p;
            sm += p;
        }
#pragma unroll
        for (int o = 1; o < 16; o <<= 1) sm += __shfl_xor(sm, o);
        rsum[j] = 1.0f / sm;
    }

    // --- P -> per-wave LDS slice (C layout -> A layout via memory) ---
    char* pbase = sP + w * 16 * 272;
#pragma unroll
    for (int nt = 0; nt < 8; ++nt)
#pragma unroll
        for (int j = 0; j < 8; ++j)
            *(bf16*)(pbase + (h * 8 + j) * 272 + (nt * 16 + lm) * 2) = (bf16)s[nt][j];
    __syncthreads();

    // --- O = P @ V' (16x64 per wave, K=128) ---
    v8f oacc[4];
#pragma unroll
    for (int nt = 0; nt < 4; ++nt) oacc[nt] = {};
#pragma unroll
    for (int kp = 0; kp < 4; ++kp) {
        FragU ap;
        const char* p = pbase + lm * 272 + kp * 64 + h * 16;
        ap.u[0] = *(const uint4*)p;
        ap.u[1] = *(const uint4*)(p + 32);
#pragma unroll
        for (int nt = 0; nt < 4; ++nt) {
            FragU bv;
            const char* q = sV + (nt * 16 + lm) * 272 + kp * 64 + h * 16;
            bv.u[0] = *(const uint4*)q;
            bv.u[1] = *(const uint4*)(q + 32);
            oacc[nt] = __builtin_amdgcn_wmma_f32_16x16x32_bf16(
                false, ap.v, false, bv.v, (short)0, oacc[nt], false, false);
        }
    }

    // --- epilogue: scale by 1/rowsum, store bf16 at (t, bj, hd*64+d) ---
#pragma unroll
    for (int nt = 0; nt < 4; ++nt) {
#pragma unroll
        for (int j = 0; j < 8; ++j) {
            const int trow = w * 16 + h * 8 + j;
            const int dloc = nt * 16 + lm;
            const float val = oacc[nt][j] * rsum[j];
            O[(size_t)trow * tstride + base + dloc] = (bf16)val;
        }
    }
}

// ---------------------------------------------------------------------------
extern "C" void kernel_launch(void* const* d_in, const int* in_sizes, int n_in,
                              void* d_out, int out_size, void* d_ws, size_t ws_size,
                              hipStream_t stream)
{
    (void)in_sizes; (void)n_in; (void)out_size; (void)ws_size;
    const float* query = (const float*)d_in[0];
    const float* key   = (const float*)d_in[1];
    const float* value = (const float*)d_in[2];
    const float* m     = (const float*)d_in[3];
    const float* ipw   = (const float*)d_in[4];
    const float* ipb   = (const float*)d_in[5];
    const float* out_w = (const float*)d_in[6];
    const float* out_b = (const float*)d_in[7];
    const float* A_q   = (const float*)d_in[8];
    const float* B_q   = (const float*)d_in[9];
    const float* A_k   = (const float*)d_in[10];
    const float* B_k   = (const float*)d_in[11];
    const float* A_v   = (const float*)d_in[12];
    const float* B_v   = (const float*)d_in[13];
    const float* A_o   = (const float*)d_in[14];
    const float* B_o   = (const float*)d_in[15];

    char* ws = (char*)d_ws;
    bf16* Wall = (bf16*)ws;                       // 4 x 512x512 bf16 = 2 MB
    bf16* Wq = Wall;
    bf16* Wk = Wall + 512 * 512;
    bf16* Wv = Wall + 2 * 512 * 512;
    bf16* Wo = Wall + 3 * 512 * 512;
    const size_t woff = (size_t)4 * 512 * 512 * 2;       // bytes
    const size_t xsz  = (size_t)M_ROWS * D_ * 2;         // 100663296 B each
    bf16* Xq = (bf16*)(ws + woff);
    bf16* Xk = (bf16*)(ws + woff + 1 * xsz);
    bf16* Xv = (bf16*)(ws + woff + 2 * xsz);
    bf16* Qb = (bf16*)(ws + woff + 3 * xsz);
    bf16* Kb = (bf16*)(ws + woff + 4 * xsz);
    bf16* Vb = (bf16*)(ws + woff + 5 * xsz);
    bf16* AO = Xq;   // Xq dead after QKV GEMMs -> reuse for attention output

    prep_weights_kernel<<<dim3(1024, 4), 256, 0, stream>>>(
        ipw, out_w, A_q, B_q, A_k, B_k, A_v, B_v, A_o, B_o, Wall);

    prep_inputs_kernel<<<dim3(49152), 256, 0, stream>>>(
        (const float4*)query, (const float4*)key, (const float4*)value,
        (const float4*)m, Xq, Xk, Xv);

    const dim3 ggrid(4, 768);  // N blocks x M blocks
    gemm_bf16_kernel<<<ggrid, 256, 0, stream>>>(Xq, Wq, ipb,        Qb, nullptr);
    gemm_bf16_kernel<<<ggrid, 256, 0, stream>>>(Xk, Wk, ipb + 512,  Kb, nullptr);
    gemm_bf16_kernel<<<ggrid, 256, 0, stream>>>(Xv, Wv, ipb + 1024, Vb, nullptr);

    attn_kernel<<<dim3(BJ_ * NH_), 256, 0, stream>>>(Qb, Kb, Vb, AO);

    gemm_bf16_kernel<<<ggrid, 256, 0, stream>>>(AO, Wo, out_b, nullptr, (float*)d_out);
}
